// WatsonCrickMultiHeadedAttention_33363305956083
// MI455X (gfx1250) — matrix operand order repaired and checked by reading, not compiled, not run on hardware
//
#include <hip/hip_runtime.h>
#include <hip/hip_bf16.h>

// MI455X / gfx1250, wave32. WMMA f16->f32 path, f16 staged operands.
typedef __attribute__((ext_vector_type(16))) _Float16 v16h;
typedef __attribute__((ext_vector_type(8)))  _Float16 v8h;
typedef __attribute__((ext_vector_type(8)))  float    v8f;

#define WMMA_F32_F16(A, B, C) \
    __builtin_amdgcn_wmma_f32_16x16x32_f16(false, (A), false, (B), (short)0, (C), false, false)

// ---------------------------------------------------------------------------
// Prep: elementwise f32 -> f16 conversion (8 elements / thread).
// ---------------------------------------------------------------------------
__global__ __launch_bounds__(256) void cvt_f16_kernel(
    const float* __restrict__ src, _Float16* __restrict__ dst, int n8)
{
    const int i = blockIdx.x * 256 + threadIdx.x;
    if (i >= n8) return;
    typedef __attribute__((ext_vector_type(8))) float vf8;
    vf8 x = ((const vf8*)src)[i];
    v8h h;
    #pragma unroll
    for (int j = 0; j < 8; ++j) h[j] = (_Float16)x[j];
    ((v8h*)dst)[i] = h;
}

// ---------------------------------------------------------------------------
// Kernel 1: row softmax of the Watson-Crick matrix. 8192 rows x 1024 cols.
// ---------------------------------------------------------------------------
__global__ __launch_bounds__(256) void wc_softmax_kernel(
    const float* __restrict__ wc, float* __restrict__ wcs)
{
    const int row  = blockIdx.x;            // 0..8191
    const int t    = threadIdx.x;
    const int wave = t >> 5;
    const int lane = t & 31;
    const float* src = wc  + (size_t)row * 1024;
    float*       dst = wcs + (size_t)row * 1024;

    __shared__ float red[8];

    float mx = -3.0e38f;
    #pragma unroll
    for (int i = 0; i < 4; ++i) mx = fmaxf(mx, src[t + i * 256]);
    #pragma unroll
    for (int off = 16; off >= 1; off >>= 1) mx = fmaxf(mx, __shfl_xor(mx, off, 32));
    if (lane == 0) red[wave] = mx;
    __syncthreads();
    {
        float m = red[lane & 7];
        #pragma unroll
        for (int off = 4; off >= 1; off >>= 1) m = fmaxf(m, __shfl_xor(m, off, 32));
        mx = m;
    }
    __syncthreads();

    float e[4];
    float sum = 0.f;
    #pragma unroll
    for (int i = 0; i < 4; ++i) { e[i] = __expf(src[t + i * 256] - mx); sum += e[i]; }
    #pragma unroll
    for (int off = 16; off >= 1; off >>= 1) sum += __shfl_xor(sum, off, 32);
    if (lane == 0) red[wave] = sum;
    __syncthreads();
    {
        float s = red[lane & 7];
        #pragma unroll
        for (int off = 4; off >= 1; off >>= 1) s += __shfl_xor(s, off, 32);
        sum = s;
    }
    const float inv = 1.0f / sum;
    #pragma unroll
    for (int i = 0; i < 4; ++i) dst[t + i * 256] = e[i] * inv;
}

// ---------------------------------------------------------------------------
// Kernel 2: y = x @ W^T + b (M=8192,K=512,N=512), pure f16 operands, f32 acc.
// TRANSPOSE=false: head-major [b][h][n][64]   (Q, K)
// TRANSPOSE=true : dk-major   [b][h][64][n]   (V, so p@V B-frags are contiguous)
// Wave -> one 16x64 output tile. grid = (8, 64), block 256.
// ---------------------------------------------------------------------------
template <bool TRANSPOSE>
__global__ __launch_bounds__(256) void proj_gemm_kernel(
    const _Float16* __restrict__ X, const _Float16* __restrict__ W,
    const float* __restrict__ bias, _Float16* __restrict__ Out, float outScale)
{
    const int lane = threadIdx.x & 31;
    const int wave = threadIdx.x >> 5;
    const bool hi  = lane >= 16;
    const int  l16 = lane & 15;
    const int  m0  = (blockIdx.y * 8 + wave) * 16;
    const int  n0  = blockIdx.x * 64;

    v8f zero = {};
    v8f acc[4];
    #pragma unroll
    for (int g = 0; g < 4; ++g) acc[g] = zero;

    #pragma unroll 1
    for (int k0 = 0; k0 < 512; k0 += 32) {
        // A fragment: lane = row; halves = K {0..15} lo-half / {16..31} hi-half
        v16h a;
        {
            const _Float16* xr = X + (size_t)(m0 + l16) * 512 + k0 + (hi ? 16 : 0);
            #pragma unroll
            for (int i = 0; i < 8; ++i) { a[i] = xr[i]; a[8 + i] = xr[8 + i]; }
        }
        #pragma unroll
        for (int g = 0; g < 4; ++g) {
            // B fragment: B[k][col] = W[col][k] -> 16 contiguous f16 from W row `col`
            const int col = n0 + g * 16 + l16;
            const _Float16* wr = W + (size_t)col * 512 + k0 + (hi ? 16 : 0);
            v16h bm;
            #pragma unroll
            for (int i = 0; i < 16; ++i) bm[i] = wr[i];
            acc[g] = WMMA_F32_F16(a, bm, acc[g]);
        }
    }

    #pragma unroll
    for (int g = 0; g < 4; ++g) {
        const int col = n0 + g * 16 + l16;
        const int h   = col >> 6;
        const int dk  = col & 63;
        const float bb = bias[col];
        #pragma unroll
        for (int r = 0; r < 8; ++r) {
            const int m = m0 + r + (hi ? 8 : 0);
            const int b = m >> 10, n = m & 1023;
            const _Float16 v = (_Float16)((acc[g][r] + bb) * outScale);
            if (TRANSPOSE)
                Out[(((size_t)b * 8 + h) * 64 + dk) * 1024 + n] = v;   // [b][h][dk][n]
            else
                Out[(((size_t)b * 8 + h) * 1024 + n) * 64 + dk] = v;   // [b][h][n][dk]
        }
    }
}

// ---------------------------------------------------------------------------
// Kernel 3: flash-attention with WC gating.
//   s = (q/sqrt(dk)) @ k^T;  s *= softmax(wc);  p = online_softmax(s);  o = p@v
// Wave -> 16 query rows; keys in chunks of 32. grid = (8, 64), block 256.
// ---------------------------------------------------------------------------
__global__ __launch_bounds__(256) void attention_kernel(
    const _Float16* __restrict__ Qh, const _Float16* __restrict__ Kh,
    const _Float16* __restrict__ Vt, const float* __restrict__ wcs,
    _Float16* __restrict__ Attn)
{
    __shared__ _Float16 lds_p[8][16 * 32];   // per-wave 16x32 p-tile transpose buffer

    const int lane = threadIdx.x & 31;
    const int wave = threadIdx.x >> 5;
    const bool hi  = lane >= 16;
    const int  l16 = lane & 15;
    const int  bh  = blockIdx.y;             // 0..63
    const int  b   = bh >> 3, h = bh & 7;
    const int  q0  = (blockIdx.x * 8 + wave) * 16;

    const _Float16* Qb  = Qh + (size_t)bh * 1024 * 64;   // [n][dk]
    const _Float16* Kb  = Kh + (size_t)bh * 1024 * 64;   // [n][dk]
    const _Float16* Vtb = Vt + (size_t)bh * 64 * 1024;   // [dk][n]
    const float* wrow   = wcs + ((size_t)b * 1024 + q0) * 1024;

    // Q A-fragments: K-dim 64 = 2 x 32
    v16h aq[2];
    #pragma unroll
    for (int kk = 0; kk < 2; ++kk) {
        const _Float16* qr = Qb + (size_t)(q0 + l16) * 64 + kk * 32 + (hi ? 16 : 0);
        #pragma unroll
        for (int i = 0; i < 8; ++i) { aq[kk][i] = qr[i]; aq[kk][8 + i] = qr[8 + i]; }
    }

    float mrun[8], lrun[8];
    v8f zero = {};
    v8f o[4];
    #pragma unroll
    for (int r = 0; r < 8; ++r) { mrun[r] = -3.0e38f; lrun[r] = 0.f; }
    #pragma unroll
    for (int g = 0; g < 4; ++g) o[g] = zero;

    #pragma unroll 1
    for (int j0 = 0; j0 < 1024; j0 += 32) {
        // ---- scores: two 16x16 tiles, K=64 via two x32 WMMAs each ----
        v8f s[2];
        #pragma unroll
        for (int t = 0; t < 2; ++t) {
            const int col = j0 + t * 16 + l16;             // key index
            const _Float16* kr = Kb + (size_t)col * 64 + (hi ? 16 : 0);
            v16h bk0, bk1;
            #pragma unroll
            for (int i = 0; i < 16; ++i) { bk0[i] = kr[i]; bk1[i] = kr[32 + i]; }
            v8f z = zero;
            z = WMMA_F32_F16(aq[0], bk0, z);
            z = WMMA_F32_F16(aq[1], bk1, z);
            s[t] = z;
        }

        // ---- gate by wc-softmax + online softmax update ----
        #pragma unroll
        for (int r = 0; r < 8; ++r) {
            const int row = r + (hi ? 8 : 0);              // row within 16-row tile
            const float* wr = wrow + (size_t)row * 1024 + j0 + l16;
            float s0 = s[0][r] * wr[0];
            float s1 = s[1][r] * wr[16];
            float cm = fmaxf(s0, s1);
            #pragma unroll
            for (int off = 8; off >= 1; off >>= 1) cm = fmaxf(cm, __shfl_xor(cm, off, 32));
            const float mnew  = fmaxf(mrun[r], cm);
            const float scale = __expf(mrun[r] - mnew);
            const float p0 = __expf(s0 - mnew);
            const float p1 = __expf(s1 - mnew);
            s[0][r] = p0; s[1][r] = p1;
            float rs = p0 + p1;
            #pragma unroll
            for (int off = 8; off >= 1; off >>= 1) rs += __shfl_xor(rs, off, 32);
            lrun[r] = lrun[r] * scale + rs;
            mrun[r] = mnew;
            #pragma unroll
            for (int g = 0; g < 4; ++g) o[g][r] *= scale;
        }

        // ---- transpose p (C/D layout -> A layout) through LDS ----
        #pragma unroll
        for (int r = 0; r < 8; ++r) {
            const int row = r + (hi ? 8 : 0);
            lds_p[wave][row * 32 + l16]      = (_Float16)s[0][r];
            lds_p[wave][row * 32 + 16 + l16] = (_Float16)s[1][r];
        }
        asm volatile("s_wait_dscnt 0x0" ::: "memory");     // in-wave LDS RAW
        v16h ap;
        {
            const _Float16* pr = &lds_p[wave][l16 * 32 + (hi ? 16 : 0)];
            #pragma unroll
            for (int i = 0; i < 8; ++i) { ap[i] = pr[i]; ap[8 + i] = pr[8 + i]; }
        }
        asm volatile("s_wait_dscnt 0x0" ::: "memory");     // loads done before next-iter stores

        // ---- o += p @ V(chunk): B[k][col] = Vt[col][j0+k] -> contiguous halves ----
        #pragma unroll
        for (int g = 0; g < 4; ++g) {
            const _Float16* vr = Vtb + (size_t)(g * 16 + l16) * 1024 + j0 + (hi ? 16 : 0);
            v16h bv;
            #pragma unroll
            for (int i = 0; i < 16; ++i) bv[i] = vr[i];
            o[g] = WMMA_F32_F16(ap, bv, o[g]);
        }
    }

    // ---- finalize: o / l, write f16 row-major [b][n][h*64+dk] for out-proj ----
    #pragma unroll
    for (int g = 0; g < 4; ++g) {
        #pragma unroll
        for (int r = 0; r < 8; ++r) {
            const int row = q0 + r + (hi ? 8 : 0);
            const float val = o[g][r] / lrun[r];
            Attn[((size_t)b * 1024 + row) * 512 + h * 64 + g * 16 + l16] = (_Float16)val;
        }
    }
}

// ---------------------------------------------------------------------------
// Kernel 4: out = attn @ Wo^T + bo, f16 operands, f32 result.
// ---------------------------------------------------------------------------
__global__ __launch_bounds__(256) void out_proj_kernel(
    const _Float16* __restrict__ A, const _Float16* __restrict__ W,
    const float* __restrict__ bias, float* __restrict__ Out)
{
    const int lane = threadIdx.x & 31;
    const int wave = threadIdx.x >> 5;
    const bool hi  = lane >= 16;
    const int  l16 = lane & 15;
    const int  m0  = (blockIdx.y * 8 + wave) * 16;
    const int  n0  = blockIdx.x * 64;

    v8f zero = {};
    v8f acc[4];
    #pragma unroll
    for (int g = 0; g < 4; ++g) acc[g] = zero;

    #pragma unroll 1
    for (int k0 = 0; k0 < 512; k0 += 32) {
        v16h a;
        {
            const _Float16* ar = A + (size_t)(m0 + l16) * 512 + k0 + (hi ? 16 : 0);
            #pragma unroll
            for (int i = 0; i < 8; ++i) { a[i] = ar[i]; a[8 + i] = ar[8 + i]; }
        }
        #pragma unroll
        for (int g = 0; g < 4; ++g) {
            const int col = n0 + g * 16 + l16;
            const _Float16* wr = W + (size_t)col * 512 + k0 + (hi ? 16 : 0);
            v16h bm;
            #pragma unroll
            for (int i = 0; i < 16; ++i) bm[i] = wr[i];
            acc[g] = WMMA_F32_F16(a, bm, acc[g]);
        }
    }

    #pragma unroll
    for (int g = 0; g < 4; ++g) {
        const int col = n0 + g * 16 + l16;
        const float bb = bias[col];
        #pragma unroll
        for (int r = 0; r < 8; ++r) {
            const int m = m0 + r + (hi ? 8 : 0);
            Out[(size_t)m * 512 + col] = acc[g][r] + bb;
        }
    }
}

// ---------------------------------------------------------------------------
// Host launcher. Inputs (setup_inputs order):
// 0 query 1 key 2 value 3 wc_matrix 4 Wq 5 bq 6 Wk 7 bk 8 Wv 9 bv 10 Wo 11 bo
// All launches on `stream` (graph-capture safe, serialized => buffer reuse OK).
// ---------------------------------------------------------------------------
extern "C" void kernel_launch(void* const* d_in, const int* in_sizes, int n_in,
                              void* d_out, int out_size, void* d_ws, size_t ws_size,
                              hipStream_t stream)
{
    const float* query = (const float*)d_in[0];
    const float* key_  = (const float*)d_in[1];
    const float* value = (const float*)d_in[2];
    const float* wc    = (const float*)d_in[3];
    const float* Wq    = (const float*)d_in[4];
    const float* bq    = (const float*)d_in[5];
    const float* Wk    = (const float*)d_in[6];
    const float* bk    = (const float*)d_in[7];
    const float* Wv    = (const float*)d_in[8];
    const float* bv    = (const float*)d_in[9];
    const float* Wo    = (const float*)d_in[10];
    const float* bo    = (const float*)d_in[11];
    float* out = (float*)d_out;

    char* ws = (char*)d_ws;
    float*    wcs = (float*)ws;                                 // 32 MB (8*1024*1024 f32)
    _Float16* Qh  = (_Float16*)(ws + (size_t)(32u << 20));      //  8 MB [b][h][n][64]
    _Float16* Kh  = (_Float16*)(ws + (size_t)(40u << 20));      //  8 MB [b][h][n][64]
    _Float16* Vt  = (_Float16*)(ws + (size_t)(48u << 20));      //  8 MB [b][h][64][n]
    _Float16* At  = (_Float16*)(ws + (size_t)(56u << 20));      //  8 MB [b][n][512]
    _Float16* Xh  = (_Float16*)(ws + (size_t)(64u << 20));      //  8 MB staged activation (reused)
    _Float16* Wh  = (_Float16*)(ws + (size_t)(72u << 20));      // 0.5 MB staged weight (reused)

    const int actN8 = (8 * 1024 * 512) / 8;   // 524288
    const int wN8   = (512 * 512) / 8;        // 32768
    const dim3 blk(256);
    const dim3 actGrid((actN8 + 255) / 256);
    const dim3 wGrid((wN8 + 255) / 256);
    const dim3 gemmGrid(8, 64);

    // 1) row softmax of wc
    wc_softmax_kernel<<<dim3(8192), blk, 0, stream>>>(wc, wcs);

    // 2) Q projection (1/sqrt(64) folded in)
    cvt_f16_kernel<<<actGrid, blk, 0, stream>>>(query, Xh, actN8);
    cvt_f16_kernel<<<wGrid,   blk, 0, stream>>>(Wq, Wh, wN8);
    proj_gemm_kernel<false><<<gemmGrid, blk, 0, stream>>>(Xh, Wh, bq, Qh, 0.125f);

    // 3) K projection
    cvt_f16_kernel<<<actGrid, blk, 0, stream>>>(key_, Xh, actN8);
    cvt_f16_kernel<<<wGrid,   blk, 0, stream>>>(Wk, Wh, wN8);
    proj_gemm_kernel<false><<<gemmGrid, blk, 0, stream>>>(Xh, Wh, bk, Kh, 1.0f);

    // 4) V projection (dk-major output)
    cvt_f16_kernel<<<actGrid, blk, 0, stream>>>(value, Xh, actN8);
    cvt_f16_kernel<<<wGrid,   blk, 0, stream>>>(Wv, Wh, wN8);
    proj_gemm_kernel<true><<<gemmGrid, blk, 0, stream>>>(Xh, Wh, bv, Vt, 1.0f);

    // 5) gated flash attention
    attention_kernel<<<gemmGrid, blk, 0, stream>>>(Qh, Kh, Vt, wcs, At);

    // 6) output projection
    cvt_f16_kernel<<<wGrid, blk, 0, stream>>>(Wo, Wh, wN8);
    out_proj_kernel<<<gemmGrid, blk, 0, stream>>>(At, Wh, bo, out);
}
